// CrossAttention_627065225372
// MI455X (gfx1250) — matrix operand order repaired and checked by reading, not compiled, not run on hardware
//
#include <hip/hip_runtime.h>

// ---------------------------------------------------------------------------
// CrossAttention with single CLS query, algebraically reduced:
//   score[b,h,n] = (Wk_h^T q_h) . x[b,n]      (skinny GEMM, bf16 WMMA)
//   out[b,h]     = Wv_h * (sum_n attn_n x[b,n])
// HBM-optimal: x (403 MB) read exactly twice -> ~35 us at 23.3 TB/s.
// ---------------------------------------------------------------------------

typedef __attribute__((ext_vector_type(16))) __bf16 v16bf;
typedef __attribute__((ext_vector_type(8)))  float  v8f;

#define B_      32
#define N_      4096
#define C_      768
#define H_      12
#define D_      64
#define NCHUNK  8
#define CHTOK   512              // tokens per chunk (N_/NCHUNK)
#define KTILES  (C_ / 32)        // 24 WMMA k-steps over C
#define SCALE_  0.125f           // 64^-0.5

// ============================ Kernel 1: prep ================================
// q[b] = Wq * x[b,0,:];  qk[b,h,c] = SCALE * sum_d Wk[h*64+d, c] * q[b,h*64+d]
// qk stored pre-swizzled in WMMA bf16 B-matrix layout:
//   qk_sw[b][kt][lane][e],  lane = ((k>>3)&1)*16 + head,  e = (k&7) + (k>>4)*8
__global__ void __launch_bounds__(256)
prep_kernel(const float* __restrict__ x, const float* __restrict__ wq,
            const float* __restrict__ wk, __bf16* __restrict__ qk_sw)
{
    __shared__ float s_xcls[C_];
    __shared__ float s_q[C_];
    const int b = blockIdx.x, tid = threadIdx.x;

    for (int c = tid; c < C_; c += 256)
        s_xcls[c] = x[(size_t)b * N_ * C_ + c];
    __syncthreads();

    for (int o = tid; o < C_; o += 256) {
        const float* wr = wq + (size_t)o * C_;
        float acc = 0.f;
        for (int c = 0; c < C_; ++c) acc += wr[c] * s_xcls[c];
        s_q[o] = acc;
    }
    __syncthreads();

    for (int idx = tid; idx < C_ * 16; idx += 256) {
        const int c = idx >> 4, h = idx & 15;
        float v = 0.f;
        if (h < H_) {
            const float* qh = &s_q[h * D_];
            for (int d = 0; d < D_; ++d)
                v += wk[(size_t)(h * D_ + d) * C_ + c] * qh[d];
            v *= SCALE_;
        }
        const int kt = c >> 5, ko = c & 31, grp = (ko >> 3) & 1, e = (ko & 7) + ((ko >> 4) << 3);
        qk_sw[((size_t)(b * KTILES + kt) * 32 + grp * 16 + h) * 16 + e] = (__bf16)v;
    }
}

// ===================== Kernel 2: fused scores + softmax + x-bar =============
__global__ void __launch_bounds__(256)
attn_chunk_kernel(const float* __restrict__ x, const __bf16* __restrict__ qk_sw,
                  float* __restrict__ ml, float* __restrict__ xbar)
{
    __shared__ float s_scores[CHTOK * 16];   // 32 KB: [token][head16]
    __shared__ float s_pmax[192];
    __shared__ float s_psum[192];
    __shared__ float s_m[16];

    const int b     = blockIdx.y;
    const int chunk = blockIdx.x;
    const int tid   = threadIdx.x;
    const int lane  = tid & 31;
    const int wave  = tid >> 5;
    const int n0    = chunk * CHTOK;

    // ---------- Pass A: scores via v_wmma_f32_16x16x32_bf16 -----------------
    const int row_in_tile = lane & 15;   // A-matrix row for this lane
    const int kgrp        = lane >> 4;   // 0: K 0..7/16..23, 1: K 8..15/24..31
    const __bf16* qb = qk_sw + ((size_t)b * KTILES * 32 + lane) * 16;

    for (int tile = wave; tile < CHTOK / 16; tile += 8) {
        const int row = n0 + tile * 16 + row_in_tile;
        const float* xr0 = x + ((size_t)b * N_ + row) * C_ + kgrp * 8;
        v8f acc = {};
        for (int kt = 0; kt < KTILES; ++kt) {
            const float* xr = xr0 + kt * 32;
            if (kt + 1 < KTILES) __builtin_prefetch(xr + 32, 0, 3);
            float4 f0 = *(const float4*)(xr);
            float4 f1 = *(const float4*)(xr + 4);
            float4 f2 = *(const float4*)(xr + 16);
            float4 f3 = *(const float4*)(xr + 20);
            v16bf a;
            a[0]=(__bf16)f0.x;  a[1]=(__bf16)f0.y;  a[2]=(__bf16)f0.z;  a[3]=(__bf16)f0.w;
            a[4]=(__bf16)f1.x;  a[5]=(__bf16)f1.y;  a[6]=(__bf16)f1.z;  a[7]=(__bf16)f1.w;
            a[8]=(__bf16)f2.x;  a[9]=(__bf16)f2.y;  a[10]=(__bf16)f2.z; a[11]=(__bf16)f2.w;
            a[12]=(__bf16)f3.x; a[13]=(__bf16)f3.y; a[14]=(__bf16)f3.z; a[15]=(__bf16)f3.w;
            v16bf bf = *(const v16bf*)(qb + (size_t)kt * 32 * 16);
            acc = __builtin_amdgcn_wmma_f32_16x16x32_bf16(
                      false, a, false, bf, (short)0, acc, false, false);
        }
        // C layout: lane L holds (M = v + (L>>4)*8, N = L&15) in VGPR v
        const int hcol = lane & 15;
        #pragma unroll
        for (int v = 0; v < 8; ++v) {
            const int m = v + kgrp * 8;
            s_scores[(tile * 16 + m) * 16 + hcol] = acc[v];
        }
    }
    __syncthreads();

    // ---------- Pass B: chunk-local softmax stats (m, l), p in LDS ----------
    if (tid < 192) {
        const int h = tid >> 4, s = tid & 15;
        float mx = -1e30f;
        for (int n = s * 32; n < s * 32 + 32; ++n)
            mx = fmaxf(mx, s_scores[n * 16 + h]);
        s_pmax[tid] = mx;
    }
    __syncthreads();
    if (tid < H_) {
        float mx = -1e30f;
        for (int s = 0; s < 16; ++s) mx = fmaxf(mx, s_pmax[tid * 16 + s]);
        s_m[tid] = mx;
    }
    __syncthreads();
    if (tid < 192) {
        const int h = tid >> 4, s = tid & 15;
        const float mh = s_m[h];
        float sum = 0.f;
        for (int n = s * 32; n < s * 32 + 32; ++n) {
            float p = __expf(s_scores[n * 16 + h] - mh);
            s_scores[n * 16 + h] = p;
            sum += p;
        }
        s_psum[tid] = sum;
    }
    __syncthreads();
    if (tid < H_) {
        float l = 0.f;
        for (int s = 0; s < 16; ++s) l += s_psum[tid * 16 + s];
        float* mlp = ml + ((size_t)(b * NCHUNK + chunk) * H_ + tid) * 2;
        mlp[0] = s_m[tid];
        mlp[1] = l;
    }
    __syncthreads();

    // ---------- Pass C: xbar_partial[h,c] = sum_n p[n,h]*x[n,c] (coalesced) -
    float acc[H_][3];
    #pragma unroll
    for (int h = 0; h < H_; ++h) { acc[h][0] = acc[h][1] = acc[h][2] = 0.f; }

    const float* xp = x + ((size_t)b * N_ + n0) * C_;
    for (int n = 0; n < CHTOK; ++n) {
        const float xv0 = xp[(size_t)n * C_ + tid];
        const float xv1 = xp[(size_t)n * C_ + tid + 256];
        const float xv2 = xp[(size_t)n * C_ + tid + 512];
        const float* pr = &s_scores[n * 16];
        #pragma unroll
        for (int h = 0; h < H_; ++h) {
            const float p = pr[h];
            acc[h][0] += p * xv0;
            acc[h][1] += p * xv1;
            acc[h][2] += p * xv2;
        }
    }
    #pragma unroll
    for (int h = 0; h < H_; ++h) {
        float* xb = xbar + ((size_t)(b * NCHUNK + chunk) * H_ + h) * C_;
        xb[tid]       = acc[h][0];
        xb[tid + 256] = acc[h][1];
        xb[tid + 512] = acc[h][2];
    }
}

// ============ Kernel 3: cross-chunk softmax merge + Wv + Wp + bias ==========
__global__ void __launch_bounds__(256)
reduce_proj_kernel(const float* __restrict__ ml, const float* __restrict__ xbar,
                   const float* __restrict__ wv, const float* __restrict__ wp,
                   const float* __restrict__ bp, float* __restrict__ out)
{
    __shared__ float s_w[NCHUNK][H_];
    __shared__ float s_xbar[H_ * C_];
    __shared__ float s_oh[C_];
    const int b = blockIdx.x, tid = threadIdx.x;

    if (tid < H_) {
        const float* mlb = ml + (size_t)b * NCHUNK * H_ * 2;
        float M = -1e30f;
        for (int c = 0; c < NCHUNK; ++c)
            M = fmaxf(M, mlb[(c * H_ + tid) * 2]);
        float L = 0.f;
        for (int c = 0; c < NCHUNK; ++c)
            L += mlb[(c * H_ + tid) * 2 + 1] * __expf(mlb[(c * H_ + tid) * 2] - M);
        const float invL = 1.f / L;
        for (int c = 0; c < NCHUNK; ++c)
            s_w[c][tid] = __expf(mlb[(c * H_ + tid) * 2] - M) * invL;
    }
    __syncthreads();

    const float* xb0 = xbar + (size_t)b * NCHUNK * H_ * C_;
    for (int idx = tid; idx < H_ * C_; idx += 256) {
        const int h = idx / C_, c = idx - h * C_;
        float acc = 0.f;
        for (int ch = 0; ch < NCHUNK; ++ch)
            acc += s_w[ch][h] * xb0[((size_t)ch * H_ + h) * C_ + c];
        s_xbar[idx] = acc;
    }
    __syncthreads();

    for (int o = tid; o < C_; o += 256) {
        const int h = o / D_;
        const float* wr = wv + (size_t)o * C_;
        const float* xh = &s_xbar[h * C_];
        float acc = 0.f;
        for (int c = 0; c < C_; ++c) acc += wr[c] * xh[c];
        s_oh[o] = acc;
    }
    __syncthreads();

    for (int o = tid; o < C_; o += 256) {
        const float* wr = wp + (size_t)o * C_;
        float acc = bp[o];
        for (int j = 0; j < C_; ++j) acc += wr[j] * s_oh[j];
        out[(size_t)b * C_ + o] = acc;
    }
}

// ============================== launch ======================================
extern "C" void kernel_launch(void* const* d_in, const int* in_sizes, int n_in,
                              void* d_out, int out_size, void* d_ws, size_t ws_size,
                              hipStream_t stream)
{
    (void)in_sizes; (void)n_in; (void)out_size; (void)ws_size;
    const float* x  = (const float*)d_in[0];
    const float* wq = (const float*)d_in[1];
    const float* wk = (const float*)d_in[2];
    const float* wv = (const float*)d_in[3];
    const float* wp = (const float*)d_in[4];
    const float* bp = (const float*)d_in[5];
    float* out = (float*)d_out;

    char* ws = (char*)d_ws;
    size_t off = 0;
    __bf16* qk_sw = (__bf16*)(ws + off);
    off += (size_t)B_ * KTILES * 32 * 16 * sizeof(__bf16);      // 768 KB
    off = (off + 255) & ~(size_t)255;
    float* ml = (float*)(ws + off);
    off += (size_t)B_ * NCHUNK * H_ * 2 * sizeof(float);        // 24 KB
    off = (off + 255) & ~(size_t)255;
    float* xbar = (float*)(ws + off);
    off += (size_t)B_ * NCHUNK * H_ * C_ * sizeof(float);       // 9.4 MB

    prep_kernel<<<B_, 256, 0, stream>>>(x, wq, wk, qk_sw);
    attn_chunk_kernel<<<dim3(NCHUNK, B_), 256, 0, stream>>>(x, qk_sw, ml, xbar);
    reduce_proj_kernel<<<B_, 256, 0, stream>>>(ml, xbar, wv, wp, bp, out);
}
